// Model_18889266168402
// MI455X (gfx1250) — matrix operand order repaired and checked by reading, not compiled, not run on hardware
//
#include <hip/hip_runtime.h>
#include <math.h>

// ---------------------------------------------------------------------------
// Types for CDNA5 WMMA (wave32, v_wmma_f32_16x16x32_bf16)
// ---------------------------------------------------------------------------
typedef __attribute__((ext_vector_type(8)))  unsigned short v8u;
typedef __attribute__((ext_vector_type(16))) unsigned short v16u;
typedef __attribute__((ext_vector_type(16))) __bf16        v16bf;
typedef __attribute__((ext_vector_type(8)))  float         v8f;

union BF16x16 { v16u u; v16bf b; };

// float -> bf16 bits, round-to-nearest-even
static __device__ __forceinline__ unsigned short f2bf(float f) {
    unsigned int u = __float_as_uint(f);
    u = (u + 0x7FFFu + ((u >> 16) & 1u)) >> 16;
    return (unsigned short)u;
}
static __device__ __forceinline__ float bf2f(unsigned short s) {
    return __uint_as_float(((unsigned int)s) << 16);
}
// monotonic float<->uint mapping for atomicMax on floats
static __device__ __forceinline__ unsigned int fmap(float f) {
    unsigned int b = __float_as_uint(f);
    return (b & 0x80000000u) ? ~b : (b | 0x80000000u);
}
static __device__ __forceinline__ float funmap(unsigned int u) {
    return __uint_as_float((u & 0x80000000u) ? (u ^ 0x80000000u) : ~u);
}

// ---------------------------------------------------------------------------
// f32 [rows,K] -> bf16 [rows,Kp] with zero pad (Kp % 32 == 0)
// ---------------------------------------------------------------------------
__global__ void k_pad_bf16(const float* __restrict__ in, unsigned short* __restrict__ out,
                           int rows, int K, int Kp) {
    long long idx = (long long)blockIdx.x * blockDim.x + threadIdx.x;
    long long total = (long long)rows * Kp;
    if (idx >= total) return;
    int r = (int)(idx / Kp);
    int k = (int)(idx % Kp);
    out[idx] = (k < K) ? f2bf(in[(long long)r * K + k]) : (unsigned short)0;
}

// ---------------------------------------------------------------------------
// WMMA GEMM: C[M,N] = A[M,Kp] * B[N,Kp]^T (+bias, optional relu)
// A,B bf16 row-major, Kp multiple of 32. Block = 256 thr = 8 waves.
// Block tile 64(M) x 128(N); each wave computes 32x32 via 2x2 WMMA tiles.
// ---------------------------------------------------------------------------
#define LSTR 40  // LDS row stride in bf16 elems (32 data + 8 pad), 16B aligned

__global__ __launch_bounds__(256)
void k_gemm_bf16(const unsigned short* __restrict__ A, const unsigned short* __restrict__ B,
                 const float* __restrict__ bias, float* __restrict__ Cf,
                 unsigned short* __restrict__ Cbf, int M, int N, int Kp, int doRelu) {
    __shared__ unsigned short sA[64 * LSTR];
    __shared__ unsigned short sB[128 * LSTR];

    const int tid  = threadIdx.x;
    const int lane = tid & 31;
    const int wave = tid >> 5;
    const int wm   = wave >> 2;   // 0..1
    const int wn   = wave & 3;    // 0..3
    const int blockM = blockIdx.y * 64;
    const int blockN = blockIdx.x * 128;

    v8f acc[2][2];
    #pragma unroll
    for (int i = 0; i < 2; ++i)
        #pragma unroll
        for (int j = 0; j < 2; ++j)
            acc[i][j] = (v8f){0.f,0.f,0.f,0.f,0.f,0.f,0.f,0.f};

    // global->LDS mapping
    const int ar = tid >> 2;          // 0..63 (A row)
    const int ac = (tid & 3) * 8;     // 0,8,16,24
    const int br = tid >> 1;          // 0..127 (B row)
    const int bc = (tid & 1) * 16;    // 0,16

    const v8u zero8 = {0,0,0,0,0,0,0,0};

    for (int k0 = 0; k0 < Kp; k0 += 32) {
        // stage A tile 64x32
        {
            int m = blockM + ar;
            v8u v = zero8;
            if (m < M) v = *(const v8u*)(A + (size_t)m * Kp + k0 + ac);
            *(v8u*)(&sA[ar * LSTR + ac]) = v;
        }
        // stage B tile 128x32 (stored as [n][k], i.e. weight-row contiguous)
        {
            int n = blockN + br;
            v8u v0 = zero8, v1 = zero8;
            if (n < N) {
                const unsigned short* p = B + (size_t)n * Kp + k0 + bc;
                v0 = *(const v8u*)(p);
                v1 = *(const v8u*)(p + 8);
            }
            *(v8u*)(&sB[br * LSTR + bc])     = v0;
            *(v8u*)(&sB[br * LSTR + bc + 8]) = v1;
        }
        __syncthreads();

        // A fragment: lanes 0-15 hold M=lane, K in {0..7,16..23}; lanes 16-31: K in {8..15,24..31}
        const int aHalf = (lane >> 4) * 8;
        const int rowA0 = wm * 32 + (lane & 15);
        BF16x16 afrag[2], bfrag[2];
        #pragma unroll
        for (int ti = 0; ti < 2; ++ti) {
            const unsigned short* pr = &sA[(rowA0 + ti * 16) * LSTR];
            v8u lo = *(const v8u*)(pr + aHalf);
            v8u hi = *(const v8u*)(pr + 16 + aHalf);
            afrag[ti].u = __builtin_shufflevector(lo, hi, 0,1,2,3,4,5,6,7,8,9,10,11,12,13,14,15);
        }
        // B fragment: lane -> col N = lane&15, K in {0..7,16..23} (lanes 0-15) / {8..15,24..31} (lanes 16-31)
        const int colB0 = wn * 32 + (lane & 15);
        const int kB    = (lane >> 4) * 8;
        #pragma unroll
        for (int tj = 0; tj < 2; ++tj) {
            const unsigned short* pr = &sB[(colB0 + tj * 16) * LSTR + kB];
            v8u lo = *(const v8u*)(pr);
            v8u hi = *(const v8u*)(pr + 16);
            bfrag[tj].u = __builtin_shufflevector(lo, hi, 0,1,2,3,4,5,6,7,8,9,10,11,12,13,14,15);
        }
        #pragma unroll
        for (int ti = 0; ti < 2; ++ti)
            #pragma unroll
            for (int tj = 0; tj < 2; ++tj)
                acc[ti][tj] = __builtin_amdgcn_wmma_f32_16x16x32_bf16(
                    false, afrag[ti].b, false, bfrag[tj].b,
                    (short)0, acc[ti][tj], false, false);
        __syncthreads();
    }

    // store: C lane layout — row = v + (lane>>4)*8, col = lane&15 within 16x16 tile
    const int crow0 = blockM + wm * 32 + (lane >> 4) * 8;
    const int ccol0 = blockN + wn * 32 + (lane & 15);
    #pragma unroll
    for (int ti = 0; ti < 2; ++ti) {
        #pragma unroll
        for (int tj = 0; tj < 2; ++tj) {
            int col = ccol0 + tj * 16;
            if (col >= N) continue;
            float bval = bias ? bias[col] : 0.f;
            #pragma unroll
            for (int v = 0; v < 8; ++v) {
                int row = crow0 + ti * 16 + v;
                if (row >= M) continue;
                float r = acc[ti][tj][v] + bval;
                if (doRelu) r = fmaxf(r, 0.f);
                if (Cf)  Cf [(size_t)row * N + col] = r;
                if (Cbf) Cbf[(size_t)row * N + col] = f2bf(r);
            }
        }
    }
}

// ---------------------------------------------------------------------------
// Graph kernels
// ---------------------------------------------------------------------------
__global__ void k_build_edges(const int* __restrict__ ei, int* __restrict__ srcL,
                              int* __restrict__ dstL, int E, int ET) {
    int i = blockIdx.x * blockDim.x + threadIdx.x;
    if (i >= ET) return;
    if (i < E) { srcL[i] = ei[i]; dstL[i] = ei[E + i]; }
    else       { srcL[i] = i - E; dstL[i] = i - E; }
}

// wave-per-(node,head): e_src[n,h] = sum_c h[n,h*C+c]*a_src[h,c]; same for dst
__global__ __launch_bounds__(256)
void k_escore(const unsigned short* __restrict__ h, const float* __restrict__ asrc,
              const float* __restrict__ adst, float* __restrict__ es,
              float* __restrict__ ed, int n, int H, int C) {
    int row  = blockIdx.x * 8 + (threadIdx.x >> 5);
    int lane = threadIdx.x & 31;
    if (row >= n * H) return;
    int nd = row / H, hh = row % H;
    int D = H * C;
    const unsigned short* hp = h + (size_t)nd * D + hh * C;
    const float* as = asrc + hh * C;
    const float* ad = adst + hh * C;
    float s1 = 0.f, s2 = 0.f;
    for (int c = lane; c < C; c += 32) {
        float v = bf2f(hp[c]);
        s1 += v * as[c];
        s2 += v * ad[c];
    }
    #pragma unroll
    for (int o = 16; o > 0; o >>= 1) {
        s1 += __shfl_xor(s1, o, 32);
        s2 += __shfl_xor(s2, o, 32);
    }
    if (lane == 0) { es[row] = s1; ed[row] = s2; }
}

__global__ void k_edge_max(const int* __restrict__ srcL, const int* __restrict__ dstL,
                           const float* __restrict__ es, const float* __restrict__ ed,
                           float* __restrict__ ebuf, unsigned int* __restrict__ mMap,
                           int ET, int H) {
    int e = blockIdx.x * blockDim.x + threadIdx.x;
    if (e >= ET) return;
    int s = srcL[e], d = dstL[e];
    for (int hh = 0; hh < H; ++hh) {
        float v = es[s * H + hh] + ed[d * H + hh];
        v = (v > 0.f) ? v : 0.2f * v;                 // leaky_relu 0.2
        ebuf[(size_t)e * H + hh] = v;
        atomicMax(&mMap[d * H + hh], fmap(v));
    }
}

__global__ void k_edge_exp(const int* __restrict__ dstL, float* __restrict__ ebuf,
                           const unsigned int* __restrict__ mMap,
                           float* __restrict__ sSum, int ET, int H) {
    int e = blockIdx.x * blockDim.x + threadIdx.x;
    if (e >= ET) return;
    int d = dstL[e];
    for (int hh = 0; hh < H; ++hh) {
        float m  = funmap(mMap[d * H + hh]);
        float ee = expf(ebuf[(size_t)e * H + hh] - m);
        ebuf[(size_t)e * H + hh] = ee;
        atomicAdd(&sSum[d * H + hh], ee);
    }
}

// thread per (edge, channel): out[dst,c] += (ee/sum) * h[src,c]
__global__ void k_aggregate(const int* __restrict__ srcL, const int* __restrict__ dstL,
                            const unsigned short* __restrict__ h,
                            const float* __restrict__ ebuf, const float* __restrict__ sSum,
                            float* __restrict__ out, int ET, int H, int C) {
    int D = H * C;
    long long idx = (long long)blockIdx.x * blockDim.x + threadIdx.x;
    long long total = (long long)ET * D;
    if (idx >= total) return;
    int e = (int)(idx / D);
    int c = (int)(idx % D);
    int hh = c / C;
    int s = srcL[e], d = dstL[e];
    float al = ebuf[(size_t)e * H + hh] / sSum[d * H + hh];
    atomicAdd(&out[(size_t)d * D + c], al * bf2f(h[(size_t)s * D + c]));
}

// out = relu(out + bias); write f32 in place + bf16 copy (stride D, D%32==0)
__global__ void k_finalize(float* __restrict__ out, const float* __restrict__ bias,
                           unsigned short* __restrict__ xbf, int n, int D) {
    long long idx = (long long)blockIdx.x * blockDim.x + threadIdx.x;
    if (idx >= (long long)n * D) return;
    int c = (int)(idx % D);
    float v = fmaxf(out[idx] + bias[c], 0.f);
    out[idx] = v;
    xbf[idx] = f2bf(v);
}

__global__ void k_pool(const float* __restrict__ feat, const int* __restrict__ batch,
                       unsigned int* __restrict__ gMap, int n, int D) {
    long long idx = (long long)blockIdx.x * blockDim.x + threadIdx.x;
    if (idx >= (long long)n * D) return;
    int nd = (int)(idx / D);
    int c  = (int)(idx % D);
    atomicMax(&gMap[batch[nd] * D + c], fmap(feat[idx]));
}

__global__ void k_unmapg(const unsigned int* __restrict__ gMap,
                         unsigned short* __restrict__ gbf, int total) {
    int i = blockIdx.x * blockDim.x + threadIdx.x;
    if (i >= total) return;
    gbf[i] = f2bf(funmap(gMap[i]));
}

// ---------------------------------------------------------------------------
// Head kernels
// ---------------------------------------------------------------------------
__global__ void k_tanh_max3(const float* a, const float* b, const float* c,
                            float* out, int total) {
    int i = blockIdx.x * blockDim.x + threadIdx.x;
    if (i >= total) return;
    out[i] = tanhf(fmaxf(fmaxf(a[i], b[i]), c[i]));
}

__global__ void k_tanh(float* a, int total) {
    int i = blockIdx.x * blockDim.x + threadIdx.x;
    if (i >= total) return;
    a[i] = tanhf(a[i]);
}

// row L2-normalize, cols == 64, wave per row
__global__ __launch_bounds__(256)
void k_norm64(const float* __restrict__ in, float* __restrict__ out, int rows) {
    int row  = blockIdx.x * 8 + (threadIdx.x >> 5);
    int lane = threadIdx.x & 31;
    if (row >= rows) return;
    float a = in[row * 64 + lane];
    float b = in[row * 64 + lane + 32];
    float ss = a * a + b * b;
    #pragma unroll
    for (int o = 16; o > 0; o >>= 1) ss += __shfl_xor(ss, o, 32);
    float inv = 1.f / fmaxf(sqrtf(ss), 1e-12f);
    out[row * 64 + lane]      = a * inv;
    out[row * 64 + lane + 32] = b * inv;
}

// freq[m,n] = 5 * dot64(dv[m], sv[n])
__global__ void k_freq(const float* __restrict__ dv, const float* __restrict__ sv,
                       float* __restrict__ out, int Bm, int Sn) {
    int i = blockIdx.x * blockDim.x + threadIdx.x;
    if (i >= Bm * Sn) return;
    int m = i / Sn, n = i % Sn;
    const float* a = dv + m * 64;
    const float* b = sv + n * 64;
    float s = 0.f;
    #pragma unroll
    for (int k = 0; k < 64; ++k) s += a[k] * b[k];
    out[i] = 5.f * s;
}

// ---------------------------------------------------------------------------
// Host driver
// ---------------------------------------------------------------------------
static inline unsigned gs(long long n) { return (unsigned)((n + 255) / 256); }
static inline size_t alup(size_t x) { return (x + 255) & ~(size_t)255; }

extern "C" void kernel_launch(void* const* d_in, const int* in_sizes, int n_in,
                              void* d_out, int out_size, void* d_ws, size_t ws_size,
                              hipStream_t stream) {
    (void)in_sizes; (void)n_in; (void)out_size; (void)ws_size;
    const int N = 40000, E = 80000, ET = 120000, NB = 1000, SE = 750;

    const float* x    = (const float*)d_in[0];
    const int*   ei   = (const int*)  d_in[1];
    const int*   batch= (const int*)  d_in[2];
    const float* win  = (const float*)d_in[3];
    const float* zin  = (const float*)d_in[4];
    const float* sein = (const float*)d_in[5];

    const int linIdx[4]  = {6, 10, 14, 18};
    const int asrcIdx[4] = {7, 11, 15, 19};
    const int adstIdx[4] = {8, 12, 16, 20};
    const int biasIdx[4] = {9, 13, 17, 21};
    const int Hs[4] = {8, 8, 8, 1};
    const int Cs[4] = {96, 128, 128, 128};
    const int Ds[4] = {768, 1024, 1024, 128};
    const int Ks[4] = {128, 768, 1024, 1024};

    // ---- workspace layout (bump allocator) ----
    char* base = (char*)d_ws;
    size_t off = 0;
    auto alloc = [&](size_t bytes) { char* p = base + off; off += alup(bytes); return p; };

    int*            srcL = (int*)           alloc(sizeof(int) * ET);
    int*            dstL = (int*)           alloc(sizeof(int) * ET);
    unsigned short* xbf  = (unsigned short*)alloc((size_t)2 * N * 1024);   // GEMM input
    unsigned short* hbf  = (unsigned short*)alloc((size_t)2 * N * 1024);   // GEMM output h
    float*          hOut = (float*)         alloc((size_t)4 * N * 1024);   // aggregation
    unsigned short* wbf  = (unsigned short*)alloc((size_t)10485760);       // weight bf16
    float*          eSrc = (float*)         alloc((size_t)4 * N * 8);
    float*          eDst = (float*)         alloc((size_t)4 * N * 8);
    unsigned int*   mMap = (unsigned int*)  alloc((size_t)4 * N * 8);
    float*          sSum = (float*)         alloc((size_t)4 * N * 8);
    float*          ebuf = (float*)         alloc((size_t)4 * ET * 8);

    // MLP scratch overlaps hbf region (free after GAT+pool)
    char* mb = (char*)hbf;
    size_t moff = 0;
    auto malloc2 = [&](size_t bytes) { char* p = mb + moff; moff += alup(bytes); return p; };
    unsigned int*   gMap   = (unsigned int*)  malloc2((size_t)4 * NB * 128);
    unsigned short* gbf    = (unsigned short*)malloc2((size_t)2 * NB * 128);
    unsigned short* g1bf   = (unsigned short*)malloc2((size_t)2 * NB * 64);
    float*          g2f    = (float*)         malloc2((size_t)4 * NB * 64);
    unsigned short* winbf  = (unsigned short*)malloc2((size_t)2 * NB * 2048);
    unsigned short* we1bf  = (unsigned short*)malloc2((size_t)2 * NB * 2048);
    float*          we2f   = (float*)         malloc2((size_t)4 * NB * 64);
    unsigned short* zinbf  = (unsigned short*)malloc2((size_t)2 * NB * 2944);
    unsigned short* ze1bf  = (unsigned short*)malloc2((size_t)2 * NB * 1600);
    float*          ze2f   = (float*)         malloc2((size_t)4 * NB * 64);
    unsigned short* sinbf  = (unsigned short*)malloc2((size_t)2 * SE * 1056);
    unsigned short* se1bf  = (unsigned short*)malloc2((size_t)2 * SE * 64);
    float*          se2f   = (float*)         malloc2((size_t)4 * SE * 64);
    float*          xwzf   = (float*)         malloc2((size_t)4 * NB * 64);

    float* outF = (float*)d_out;
    float* dv   = outF + (size_t)NB * SE;           // 750000
    float* sv   = dv + (size_t)NB * 64;             // +64000

    // ---- graph setup ----
    k_build_edges<<<gs(ET), 256, 0, stream>>>(ei, srcL, dstL, E, ET);
    k_pad_bf16<<<gs((long long)N * 128), 256, 0, stream>>>(x, xbf, N, 128, 128);

    // ---- 4 GAT layers ----
    for (int l = 0; l < 4; ++l) {
        const int H = Hs[l], C = Cs[l], D = Ds[l], K = Ks[l];
        const float* linW = (const float*)d_in[linIdx[l]];
        const float* aS   = (const float*)d_in[asrcIdx[l]];
        const float* aD   = (const float*)d_in[adstIdx[l]];
        const float* bb   = (const float*)d_in[biasIdx[l]];

        // weight f32 -> bf16 (K already multiple of 32)
        k_pad_bf16<<<gs((long long)D * K), 256, 0, stream>>>(linW, wbf, D, K, K);
        // h = x @ W^T  (bf16 out only)
        dim3 grid((D + 127) / 128, (N + 63) / 64);
        k_gemm_bf16<<<grid, 256, 0, stream>>>(xbf, wbf, nullptr, nullptr, hbf, N, D, K, 0);

        // attention scores
        k_escore<<<gs((long long)N * H * 32) , 256, 0, stream>>>(hbf, aS, aD, eSrc, eDst, N, H, C);
        hipMemsetAsync(mMap, 0, sizeof(unsigned int) * N * H, stream);
        hipMemsetAsync(sSum, 0, sizeof(float) * N * H, stream);
        k_edge_max<<<gs(ET), 256, 0, stream>>>(srcL, dstL, eSrc, eDst, ebuf, mMap, ET, H);
        k_edge_exp<<<gs(ET), 256, 0, stream>>>(dstL, ebuf, mMap, sSum, ET, H);

        // aggregation
        hipMemsetAsync(hOut, 0, (size_t)4 * N * D, stream);
        k_aggregate<<<gs((long long)ET * D), 256, 0, stream>>>(srcL, dstL, hbf, ebuf, sSum,
                                                               hOut, ET, H, C);
        // bias + relu + bf16 for next layer
        k_finalize<<<gs((long long)N * D), 256, 0, stream>>>(hOut, bb, xbf, N, D);
    }

    // ---- global max pool (hOut holds relu'd layer-4 features [N,128]) ----
    hipMemsetAsync(gMap, 0, sizeof(unsigned int) * NB * 128, stream);
    k_pool<<<gs((long long)N * 128), 256, 0, stream>>>(hOut, batch, gMap, N, 128);
    k_unmapg<<<gs(NB * 128), 256, 0, stream>>>(gMap, gbf, NB * 128);

    // ---- MLP head ----
    auto gemm = [&](const unsigned short* A, const unsigned short* B, const float* bias,
                    float* Cf, unsigned short* Cbf, int M, int Nn, int Kp, int relu) {
        dim3 grid((Nn + 127) / 128, (M + 63) / 64);
        k_gemm_bf16<<<grid, 256, 0, stream>>>(A, B, bias, Cf, Cbf, M, Nn, Kp, relu);
    };
    auto conv = [&](const float* W, int rows, int K, int Kp) {
        k_pad_bf16<<<gs((long long)rows * Kp), 256, 0, stream>>>(W, wbf, rows, K, Kp);
    };

    // g path: [1000,128] -> 64 -> 64
    conv((const float*)d_in[22], 64, 128, 128);
    gemm(gbf, wbf, (const float*)d_in[23], nullptr, g1bf, NB, 64, 128, 1);
    conv((const float*)d_in[24], 64, 64, 64);
    gemm(g1bf, wbf, (const float*)d_in[25], g2f, nullptr, NB, 64, 64, 1);

    // w path: [1000,2048] -> 2048 -> 64
    k_pad_bf16<<<gs((long long)NB * 2048), 256, 0, stream>>>(win, winbf, NB, 2048, 2048);
    conv((const float*)d_in[26], 2048, 2048, 2048);
    gemm(winbf, wbf, (const float*)d_in[27], nullptr, we1bf, NB, 2048, 2048, 1);
    conv((const float*)d_in[28], 64, 2048, 2048);
    gemm(we1bf, wbf, (const float*)d_in[29], we2f, nullptr, NB, 64, 2048, 1);

    // z path: [1000,2916] -> 1600 -> 64  (pad K 2916 -> 2944)
    k_pad_bf16<<<gs((long long)NB * 2944), 256, 0, stream>>>(zin, zinbf, NB, 2916, 2944);
    conv((const float*)d_in[30], 1600, 2916, 2944);
    gemm(zinbf, wbf, (const float*)d_in[31], nullptr, ze1bf, NB, 1600, 2944, 1);
    conv((const float*)d_in[32], 64, 1600, 1600);
    gemm(ze1bf, wbf, (const float*)d_in[33], ze2f, nullptr, NB, 64, 1600, 1);

    // side-effects path: [750,1050] -> 64 -> 64 (pad 1050 -> 1056)
    k_pad_bf16<<<gs((long long)SE * 1056), 256, 0, stream>>>(sein, sinbf, SE, 1050, 1056);
    conv((const float*)d_in[34], 64, 1050, 1056);
    gemm(sinbf, wbf, (const float*)d_in[35], nullptr, se1bf, SE, 64, 1056, 1);
    conv((const float*)d_in[36], 64, 64, 64);
    gemm(se1bf, wbf, (const float*)d_in[37], se2f, nullptr, SE, 64, 64, 0);

    // combine + normalize + similarity
    k_tanh_max3<<<gs(NB * 64), 256, 0, stream>>>(g2f, we2f, ze2f, xwzf, NB * 64);
    k_tanh<<<gs(SE * 64), 256, 0, stream>>>(se2f, SE * 64);
    k_norm64<<<gs((long long)NB * 32), 256, 0, stream>>>(xwzf, dv, NB);
    k_norm64<<<gs((long long)SE * 32), 256, 0, stream>>>(se2f, sv, SE);
    k_freq<<<gs((long long)NB * SE), 256, 0, stream>>>(dv, sv, outF, NB, SE);
}
